// Method_GNN_Cora_2680059592878
// MI455X (gfx1250) — matrix operand order repaired and checked by reading, not compiled, not run on hardware
//
#include <hip/hip_runtime.h>
#include <hip/hip_bf16.h>
#include <math.h>

typedef _Float16 v16h __attribute__((ext_vector_type(16)));
typedef float    v8f  __attribute__((ext_vector_type(8)));
typedef float    v4f  __attribute__((ext_vector_type(4)));

#define NNODES 100000
#define NEDGES 3200000
#define KDIM   1433
#define KPAD   1440
#define HID    40
#define HPAD   48
#define NCLS   7
#define CPAD   8

// ---------------------------------------------------------------------------
// Prep: W1 (1433x40, row-major f32) -> W1t (48 cols x 1440 k, col-major f16,
// zero padded). Makes B-fragment loads for WMMA fully contiguous & aligned.
// ---------------------------------------------------------------------------
__global__ void prep_w1t(const float* __restrict__ W1, _Float16* __restrict__ w1t) {
  int idx = blockIdx.x * blockDim.x + threadIdx.x;
  if (idx >= HPAD * KPAD) return;
  int n = idx / KPAD;
  int k = idx - n * KPAD;
  float v = (n < HID && k < KDIM) ? W1[(size_t)k * HID + n] : 0.0f;
  w1t[idx] = (_Float16)v;
}

__global__ void zero_f32(float* __restrict__ p, int n) {
  int i = blockIdx.x * blockDim.x + threadIdx.x;
  if (i < n) p[i] = 0.0f;
}

// ---------------------------------------------------------------------------
// GEMM1: support1[N x 40] = x[N x 1433] @ W1.  WMMA f32<-f16, 16x16x32.
// Block: 256 threads = 8 waves; each wave owns a 16-row x 48-col tile
// (3 accumulators across the padded 48 columns); block covers 128 rows.
// A layout (matches B): lane<16 holds K {kb..kb+15},
// lane>=16 holds K {kb+16..kb+31} of row M=lane%16.
// B layout: lane holds column N=lane%16, K run of 16 ((lane/16)*16 offset).
// ---------------------------------------------------------------------------
__global__ __launch_bounds__(256) void gemm1_wmma(
    const float* __restrict__ x,
    const _Float16* __restrict__ w1t,
    float* __restrict__ support1)
{
  const int lane = threadIdx.x & 31;
  const int wave = threadIdx.x >> 5;
  const int l16  = lane & 15;
  const int half = lane >> 4;               // 0 or 1
  const int rowBase = blockIdx.x * 128 + wave * 16;

  int arow = rowBase + l16;
  if (arow >= NNODES) arow = NNODES - 1;    // clamp reads; stores are guarded
  const float* __restrict__ xrow = x + (size_t)arow * KDIM;

  const _Float16* __restrict__ b0p = w1t + (size_t)( 0 + l16) * KPAD + half * 16;
  const _Float16* __restrict__ b1p = w1t + (size_t)(16 + l16) * KPAD + half * 16;
  const _Float16* __restrict__ b2p = w1t + (size_t)(32 + l16) * KPAD + half * 16;

  v8f c0 = {}, c1 = {}, c2 = {};

  int kb = 0;
  // main K loop: 44 full 32-wide steps (covers K=0..1407)
  for (; kb + 32 <= KDIM; kb += 32) {
    const float* p0 = xrow + kb + half * 16;
    const float* p1 = xrow + kb + half * 16 + 8;
    v16h a;
#pragma unroll
    for (int i = 0; i < 8; ++i) a[i]     = (_Float16)p0[i];
#pragma unroll
    for (int i = 0; i < 8; ++i) a[8 + i] = (_Float16)p1[i];

    v16h b0 = *(const v16h*)(b0p + kb);
    v16h b1 = *(const v16h*)(b1p + kb);
    v16h b2 = *(const v16h*)(b2p + kb);

    c0 = __builtin_amdgcn_wmma_f32_16x16x32_f16(false, a, false, b0, (short)0, c0, false, false);
    c1 = __builtin_amdgcn_wmma_f32_16x16x32_f16(false, a, false, b1, (short)0, c1, false, false);
    c2 = __builtin_amdgcn_wmma_f32_16x16x32_f16(false, a, false, b2, (short)0, c2, false, false);
  }
  // K tail (kb = 1408, 25 valid k); W1t is zero-padded so only A needs guards
  {
    v16h a;
#pragma unroll
    for (int i = 0; i < 8; ++i) {
      int k0 = kb + half * 16 + i;
      int k1 = kb + half * 16 + 8 + i;
      a[i]     = (k0 < KDIM) ? (_Float16)xrow[k0] : (_Float16)0.0f;
      a[8 + i] = (k1 < KDIM) ? (_Float16)xrow[k1] : (_Float16)0.0f;
    }
    v16h b0 = *(const v16h*)(b0p + kb);
    v16h b1 = *(const v16h*)(b1p + kb);
    v16h b2 = *(const v16h*)(b2p + kb);
    c0 = __builtin_amdgcn_wmma_f32_16x16x32_f16(false, a, false, b0, (short)0, c0, false, false);
    c1 = __builtin_amdgcn_wmma_f32_16x16x32_f16(false, a, false, b1, (short)0, c1, false, false);
    c2 = __builtin_amdgcn_wmma_f32_16x16x32_f16(false, a, false, b2, (short)0, c2, false, false);
  }

  // C/D layout: VGPR i, lane L -> row M = i + 8*(L/16), col N = L%16 (+tile*16)
#pragma unroll
  for (int i = 0; i < 8; ++i) {
    int row = rowBase + i + half * 8;
    if (row < NNODES) {
      float* o = support1 + (size_t)row * HID;
      o[l16]      = c0[i];
      o[16 + l16] = c1[i];
      if (l16 < 8) o[32 + l16] = c2[i];
    }
  }
}

// ---------------------------------------------------------------------------
// SpMM layer 1: agg1[dst] += support1[src] * w, 40 feats/edge.
// support1/agg1 are L2-resident (16 MB each); HW f32 atomics, agent scope.
// ---------------------------------------------------------------------------
__global__ void spmm40(const int* __restrict__ src, const int* __restrict__ dst,
                       const float* __restrict__ ew,
                       const float* __restrict__ sup, float* __restrict__ agg,
                       int nedge)
{
  int e = blockIdx.x * blockDim.x + threadIdx.x;
  if (e >= nedge) return;
  int s = src[e], d = dst[e];
  float w = ew[e];
  const v4f* __restrict__ sp = (const v4f*)(sup + (size_t)s * HID);  // 160B rows, 32B aligned
  float* __restrict__ ag = agg + (size_t)d * HID;
#pragma unroll
  for (int q = 0; q < 10; ++q) {
    v4f v = sp[q];
#pragma unroll
    for (int j = 0; j < 4; ++j)
      __hip_atomic_fetch_add(ag + q * 4 + j, v[j] * w,
                             __ATOMIC_RELAXED, __HIP_MEMORY_SCOPE_AGENT);
  }
}

// bias + ReLU + dropout mask (keep if u > 0.5, scale by 1/(1-p) = 2.0), in place
__global__ void relu_drop(float* __restrict__ h, const float* __restrict__ b1,
                          const float* __restrict__ mask, int n)
{
  int i = blockIdx.x * blockDim.x + threadIdx.x;
  if (i >= n) return;
  float v = h[i] + b1[i % HID];
  v = v > 0.0f ? v : 0.0f;
  h[i] = (mask[i] > 0.5f) ? v * 2.0f : 0.0f;
}

// GEMM2: support2[N x 7 (stride 8)] = h[N x 40] @ W2[40 x 7]  (tiny; scalar f32)
__global__ void gemm2(const float* __restrict__ h, const float* __restrict__ W2,
                      float* __restrict__ sup2)
{
  int row = blockIdx.x * blockDim.x + threadIdx.x;
  if (row >= NNODES) return;
  const float* hr = h + (size_t)row * HID;
  float acc[NCLS] = {};
#pragma unroll 8
  for (int k = 0; k < HID; ++k) {
    float hv = hr[k];
#pragma unroll
    for (int c = 0; c < NCLS; ++c) acc[c] += hv * W2[k * NCLS + c];
  }
  float* o = sup2 + (size_t)row * CPAD;
#pragma unroll
  for (int c = 0; c < NCLS; ++c) o[c] = acc[c];
  o[NCLS] = 0.0f;
}

// SpMM layer 2: agg2[dst] += support2[src] * w, 7 feats/edge (stride 8)
__global__ void spmm7(const int* __restrict__ src, const int* __restrict__ dst,
                      const float* __restrict__ ew,
                      const float* __restrict__ sup2, float* __restrict__ agg2,
                      int nedge)
{
  int e = blockIdx.x * blockDim.x + threadIdx.x;
  if (e >= nedge) return;
  int s = src[e], d = dst[e];
  float w = ew[e];
  const float* __restrict__ sp = sup2 + (size_t)s * CPAD;
  float* __restrict__ ag = agg2 + (size_t)d * CPAD;
#pragma unroll
  for (int c = 0; c < NCLS; ++c)
    __hip_atomic_fetch_add(ag + c, sp[c] * w,
                           __ATOMIC_RELAXED, __HIP_MEMORY_SCOPE_AGENT);
}

// bias + log_softmax over 7 classes per row -> d_out (N x 7)
__global__ void lsm7(const float* __restrict__ agg2, const float* __restrict__ b2,
                     float* __restrict__ out)
{
  int row = blockIdx.x * blockDim.x + threadIdx.x;
  if (row >= NNODES) return;
  const float* a = agg2 + (size_t)row * CPAD;
  float v[NCLS];
  float m = -INFINITY;
#pragma unroll
  for (int c = 0; c < NCLS; ++c) { v[c] = a[c] + b2[c]; m = fmaxf(m, v[c]); }
  float s = 0.0f;
#pragma unroll
  for (int c = 0; c < NCLS; ++c) s += __expf(v[c] - m);
  float lse = m + __logf(s);
  float* o = out + (size_t)row * NCLS;
#pragma unroll
  for (int c = 0; c < NCLS; ++c) o[c] = v[c] - lse;
}

extern "C" void kernel_launch(void* const* d_in, const int* in_sizes, int n_in,
                              void* d_out, int out_size, void* d_ws, size_t ws_size,
                              hipStream_t stream)
{
  const float* x    = (const float*)d_in[0];
  const int*   src  = (const int*)  d_in[1];
  const int*   dst  = (const int*)  d_in[2];
  const float* ew   = (const float*)d_in[3];
  const float* W1   = (const float*)d_in[4];
  const float* b1   = (const float*)d_in[5];
  const float* W2   = (const float*)d_in[6];
  const float* b2   = (const float*)d_in[7];
  const float* mask = (const float*)d_in[8];
  float* out = (float*)d_out;

  const int nedge = in_sizes[1];            // 3200000

  // workspace layout (floats): support1 | support2 | agg1 | agg2 | w1t(f16)
  float* ws       = (float*)d_ws;
  float* support1 = ws;                                   // N*40  = 4,000,000
  float* support2 = support1 + (size_t)NNODES * HID;      // N*8   =   800,000
  float* agg1     = support2 + (size_t)NNODES * CPAD;     // N*40  = 4,000,000
  float* agg2     = agg1     + (size_t)NNODES * HID;      // N*8   =   800,000
  _Float16* w1t   = (_Float16*)(agg2 + (size_t)NNODES * CPAD); // 48*1440 halves

  const int zero_n = NNODES * HID + NNODES * CPAD;        // agg1+agg2 contiguous

  prep_w1t  <<<(HPAD * KPAD + 255) / 256, 256, 0, stream>>>(W1, w1t);
  zero_f32  <<<(zero_n + 255) / 256,      256, 0, stream>>>(agg1, zero_n);
  gemm1_wmma<<<(NNODES + 127) / 128,      256, 0, stream>>>(x, w1t, support1);
  spmm40    <<<(nedge + 255) / 256,       256, 0, stream>>>(src, dst, ew, support1, agg1, nedge);
  relu_drop <<<(NNODES * HID + 255) / 256,256, 0, stream>>>(agg1, b1, mask, NNODES * HID);
  gemm2     <<<(NNODES + 255) / 256,      256, 0, stream>>>(agg1, W2, support2);
  spmm7     <<<(nedge + 255) / 256,       256, 0, stream>>>(src, dst, ew, support2, agg2, nedge);
  lsm7      <<<(NNODES + 255) / 256,      256, 0, stream>>>(agg2, b2, out);
}
